// MultiHeadEncoder_33423435497581
// MI455X (gfx1250) — matrix operand order repaired and checked by reading, not compiled, not run on hardware
//
#include <hip/hip_runtime.h>
#include <hip/hip_bf16.h>
#include <stdint.h>

// ---------------------------------------------------------------------------
// MI455X (gfx1250) implementation of the 3-head dense-emulated spconv stack.
// bf16 activations/weights, implicit GEMM via v_wmma_f32_16x16x32_bf16 with
// f32 accumulation. Weight B-fragments are cooperatively staged through LDS
// (double-buffered, one workgroup barrier per K-chunk) since they are shared
// by all 8 waves of a block. Fused BN+ReLU+occupancy-mask epilogue.
// ---------------------------------------------------------------------------

typedef __attribute__((ext_vector_type(16))) __bf16 v16bf;
typedef __attribute__((ext_vector_type(8)))  float  v8f;

#define WAVES_PER_BLOCK 8

// ---- small helper kernels --------------------------------------------------

__global__ void k_f32_to_bf16(const float* __restrict__ in,
                              __hip_bfloat16* __restrict__ out, int n) {
  int i = blockIdx.x * blockDim.x + threadIdx.x;
  if (i < n) out[i] = __float2bfloat16(in[i]);
}

__global__ void k_mask_init(const float* __restrict__ m,
                            unsigned char* __restrict__ out, int n) {
  int i = blockIdx.x * blockDim.x + threadIdx.x;
  if (i < n) out[i] = (m[i] > 0.5f) ? 1 : 0;
}

// new active set for stride-2 spconv: OR over the 27 strided taps
__global__ void k_mask_down(const unsigned char* __restrict__ mi,
                            unsigned char* __restrict__ mo,
                            int Din, int Hin, int Win,
                            int Dout, int Hout, int Wout,
                            int sd, int sh, int sw, int pd, int ph, int pw) {
  int i = blockIdx.x * blockDim.x + threadIdx.x;
  int total = Dout * Hout * Wout;
  if (i >= total) return;
  int w = i % Wout, h = (i / Wout) % Hout, d = i / (Wout * Hout);
  int any = 0;
  for (int td = 0; td < 3; ++td) {
    int di = d * sd - pd + td; if ((unsigned)di >= (unsigned)Din) continue;
    for (int th = 0; th < 3; ++th) {
      int hi = h * sh - ph + th; if ((unsigned)hi >= (unsigned)Hin) continue;
      for (int tw = 0; tw < 3; ++tw) {
        int wi = w * sw - pw + tw; if ((unsigned)wi >= (unsigned)Win) continue;
        any |= mi[(di * Hin + hi) * Win + wi];
      }
    }
  }
  mo[i] = any ? 1 : 0;
}

// Pack f32 weights (3,3,3,Cin,Cout) into the exact WMMA B-fragment lane layout:
// frag[((tileN*nChunks + chunk)*32 + lane)*16 + e]  holds  B[K][N] with
//   N = (lane&15) + 16*tileN ,  K = chunk*32 + 16*(e>>3) + 8*(lane>>4) + (e&7).
// Taps >= 27 (K padding) are zero-filled.
__global__ void k_pack_w(const float* __restrict__ w,
                         __hip_bfloat16* __restrict__ frag,
                         int Cin, int Cout, int nChunks) {
  int i = blockIdx.x * blockDim.x + threadIdx.x;
  int total = (Cout >> 4) * nChunks * 512;
  if (i >= total) return;
  int e     = i & 15;
  int lane  = (i >> 4) & 31;
  int rest  = i >> 9;
  int chunk = rest % nChunks;
  int tileN = rest / nChunks;
  int g = lane >> 4;
  int n = (lane & 15) + tileN * 16;
  int K = chunk * 32 + (e >> 3) * 16 + g * 8 + (e & 7);
  int tap = K / Cin, cin = K % Cin;
  float v = 0.f;
  if (tap < 27) v = w[(tap * Cin + cin) * Cout + n];
  frag[i] = __float2bfloat16(v);
}

// ---- main WMMA implicit-GEMM conv layer ------------------------------------
// One wave computes 16 output voxels (contiguous along W) x all COUT channels.
// A fragment (bf16 16x32): lane L holds row M=L&15, elements form two 8-K runs
//   run r: K = ch*32 + 16*r + 8*g + {0..7}  (g = L>>4)
// which map to contiguous input channels -> one b128 load per run (2x b64 for
// CIN=4). B fragments (shared by all waves of the block) are staged through
// LDS, double-buffered, one barrier per chunk.
template <int CIN, int COUT, bool FINAL>
__global__ __launch_bounds__(256) void k_conv_wmma(
    const __hip_bfloat16* __restrict__ in,    // [Din*Hin*Win*CIN] bf16
    const __hip_bfloat16* __restrict__ wfrag, // packed B fragments
    const float* __restrict__ scale,
    const float* __restrict__ bias,
    const unsigned char* __restrict__ mask,   // output-resolution occupancy
    __hip_bfloat16* __restrict__ out,         // bf16 activations (non-final)
    float* __restrict__ outf,                 // f32 head-mean accumulator (final)
    float headW, int accumAdd,
    int Din, int Hin, int Win, int Dout, int Hout, int Wout,
    int sd, int sh, int sw, int pd, int ph, int pw) {
  constexpr int NCH = (27 * CIN + 31) / 32;  // K chunks of 32
  constexpr int NCT = COUT / 16;             // N tiles

  // double-buffered B-fragment staging: 2 slots x NCT x 256 dwords
  __shared__ uint32_t ldsb[2 * NCT * 256];

  const int tid = threadIdx.x;
  const int lane = tid & 31;
  const int wave = blockIdx.x * WAVES_PER_BLOCK + (tid >> 5);
  const int nTW = (Wout + 15) >> 4;
  const int totalTiles = Dout * Hout * nTW;
  const bool active = (wave < totalTiles);  // wave-uniform

  const int wt = wave % nTW;
  const int dh = wave / nTW;
  const int h = dh % Hout;
  const int d = dh / Hout;
  const int w0 = wt << 4;

  const int g = lane >> 4;
  const int row = lane & 15;  // A-matrix row M for this lane

  const int dbase = d * sd - pd;
  const int hbase = h * sh - ph;
  const int wbase = (w0 + row) * sw - pw;

  const uint32_t* __restrict__ wf32 = (const uint32_t*)wfrag;

  // cooperative stage of one chunk's B fragments (NCT KB) into an LDS slot
  auto stage = [&](int ch) {
    const int base = (ch & 1) * (NCT * 256);
#pragma unroll
    for (int t = 0; t < NCT; ++t)
      ldsb[base + t * 256 + tid] = wf32[((size_t)t * NCH + ch) * 256 + tid];
  };

  v8f zero = {};
  v8f acc[NCT];
#pragma unroll
  for (int t = 0; t < NCT; ++t) acc[t] = zero;

  stage(0);

  for (int ch = 0; ch < NCH; ++ch) {
    __syncthreads();                       // staged slot visible; old reads done
    if (ch + 1 < NCH) stage(ch + 1);       // prefetch next slot (disjoint)

    if (active) {
      uint32_t areg[8];
#pragma unroll
      for (int r = 0; r < 2; ++r) {
        const int kbase = ch * 32 + r * 16 + g * 8;
        if constexpr (CIN >= 8) {
          const int tap = kbase / CIN;
          const int c0 = kbase % CIN;
          uint4 v = make_uint4(0u, 0u, 0u, 0u);
          if (tap < 27) {
            const int td = tap / 9, th = (tap / 3) % 3, tw = tap % 3;
            const int di = dbase + td, hi = hbase + th, wi = wbase + tw;
            if ((unsigned)di < (unsigned)Din && (unsigned)hi < (unsigned)Hin &&
                (unsigned)wi < (unsigned)Win) {
              const size_t vox = ((size_t)di * Hin + hi) * Win + wi;
              v = *(const uint4*)(in + vox * CIN + c0);  // 8 contiguous bf16
            }
          }
          areg[r * 4 + 0] = v.x; areg[r * 4 + 1] = v.y;
          areg[r * 4 + 2] = v.z; areg[r * 4 + 3] = v.w;
        } else {  // CIN == 4: an 8-K run spans two taps of 4 channels each
#pragma unroll
          for (int s = 0; s < 2; ++s) {
            const int kb = kbase + s * 4;
            const int tap = kb >> 2;
            uint2 v = make_uint2(0u, 0u);
            if (tap < 27) {
              const int td = tap / 9, th = (tap / 3) % 3, tw = tap % 3;
              const int di = dbase + td, hi = hbase + th, wi = wbase + tw;
              if ((unsigned)di < (unsigned)Din && (unsigned)hi < (unsigned)Hin &&
                  (unsigned)wi < (unsigned)Win) {
                const size_t vox = ((size_t)di * Hin + hi) * Win + wi;
                v = *(const uint2*)(in + vox * 4);  // 4 contiguous bf16
              }
            }
            areg[r * 4 + s * 2 + 0] = v.x;
            areg[r * 4 + s * 2 + 1] = v.y;
          }
        }
      }
      v16bf a;
      __builtin_memcpy(&a, areg, sizeof(a));

      const int sbase = (ch & 1) * (NCT * 256) + lane * 8;
#pragma unroll
      for (int t = 0; t < NCT; ++t) {
        const uint32_t* bp = &ldsb[sbase + t * 256];
        uint32_t breg[8];
        const uint4 b0 = *(const uint4*)(bp);      // ds_load_b128
        const uint4 b1 = *(const uint4*)(bp + 4);  // ds_load_b128
        breg[0] = b0.x; breg[1] = b0.y; breg[2] = b0.z; breg[3] = b0.w;
        breg[4] = b1.x; breg[5] = b1.y; breg[6] = b1.z; breg[7] = b1.w;
        v16bf b;
        __builtin_memcpy(&b, breg, sizeof(b));
        acc[t] = __builtin_amdgcn_wmma_f32_16x16x32_bf16(
            false, a, false, b, (short)0, acc[t], false, false);
      }
    }
  }

  if (!active) return;

  // Epilogue: C/D layout lane L -> N = L&15, VGPR r -> M = r + 8*g.
  const int n0 = lane & 15;
  float sc[NCT], bi[NCT];
#pragma unroll
  for (int t = 0; t < NCT; ++t) {
    sc[t] = scale[t * 16 + n0];
    bi[t] = bias[t * 16 + n0];
  }
  const size_t rowBase = ((size_t)d * Hout + h) * Wout;
#pragma unroll
  for (int r = 0; r < 8; ++r) {
    const int M = r + 8 * g;
    const int w = w0 + M;
    if (w >= Wout) continue;
    const float mval = mask[rowBase + w] ? 1.f : 0.f;
#pragma unroll
    for (int t = 0; t < NCT; ++t) {
      float v = acc[t][r] * sc[t] + bi[t];
      v = (v > 0.f ? v : 0.f) * mval;
      const size_t oidx = (rowBase + w) * COUT + (t * 16 + n0);
      if constexpr (FINAL) {
        const float c = v * headW;
        if (accumAdd) outf[oidx] += c; else outf[oidx] = c;
      } else {
        out[oidx] = __float2bfloat16(v);
      }
    }
  }
}

// ---- host orchestration ----------------------------------------------------

extern "C" void kernel_launch(void* const* d_in, const int* in_sizes, int n_in,
                              void* d_out, int out_size, void* d_ws,
                              size_t ws_size, hipStream_t stream) {
  static const int LVL[4][3] = {{41, 256, 224}, {21, 128, 112}, {11, 64, 56}, {5, 32, 28}};
  struct LD { int cin, cout, li, lo, sd, sh, sw, pd, ph, pw; };
  static const LD L[11] = {
      {4, 16, 0, 0, 1, 1, 1, 1, 1, 1},   // 0 subm
      {16, 16, 0, 0, 1, 1, 1, 1, 1, 1},  // 1 subm
      {16, 32, 0, 1, 2, 2, 2, 1, 1, 1},  // 2 spconv
      {32, 32, 1, 1, 1, 1, 1, 1, 1, 1},  // 3 subm
      {32, 32, 1, 1, 1, 1, 1, 1, 1, 1},  // 4 subm
      {32, 64, 1, 2, 2, 2, 2, 1, 1, 1},  // 5 spconv
      {64, 64, 2, 2, 1, 1, 1, 1, 1, 1},  // 6 subm
      {64, 64, 2, 2, 1, 1, 1, 1, 1, 1},  // 7 subm
      {64, 64, 2, 3, 2, 2, 2, 0, 1, 1},  // 8 spconv (D pad 0)
      {64, 64, 3, 3, 1, 1, 1, 1, 1, 1},  // 9 subm
      {64, 64, 3, 3, 1, 1, 1, 1, 1, 1},  // 10 subm
  };
  const float* x = (const float*)d_in[0];
  const float* mf = (const float*)d_in[1];

  char* ws = (char*)d_ws;
  size_t off = 0;
  auto carve = [&](size_t b) -> void* {
    off = (off + 255) & ~(size_t)255;
    void* p = ws + off;
    off += b;
    return p;
  };
  size_t nvox[4];
  for (int i = 0; i < 4; ++i)
    nvox[i] = (size_t)LVL[i][0] * LVL[i][1] * LVL[i][2];

  __hip_bfloat16* xbf = (__hip_bfloat16*)carve(nvox[0] * 4 * sizeof(__hip_bfloat16));
  unsigned char* mk[4];
  for (int i = 0; i < 4; ++i) mk[i] = (unsigned char*)carve(nvox[i]);
  __hip_bfloat16* actA = (__hip_bfloat16*)carve(nvox[0] * 16 * sizeof(__hip_bfloat16));
  __hip_bfloat16* actB = (__hip_bfloat16*)carve(nvox[0] * 16 * sizeof(__hip_bfloat16));
  __hip_bfloat16* wfrag = (__hip_bfloat16*)carve((size_t)4 * 54 * 512 * sizeof(__hip_bfloat16));

  {
    int n = (int)(nvox[0] * 4);
    k_f32_to_bf16<<<(n + 255) / 256, 256, 0, stream>>>(x, xbf, n);
  }
  {
    int n = (int)nvox[0];
    k_mask_init<<<(n + 255) / 256, 256, 0, stream>>>(mf, mk[0], n);
  }
  const int mlayers[3] = {2, 5, 8};
  for (int j = 0; j < 3; ++j) {
    const LD& ld = L[mlayers[j]];
    int n = (int)nvox[ld.lo];
    k_mask_down<<<(n + 255) / 256, 256, 0, stream>>>(
        mk[ld.li], mk[ld.lo], LVL[ld.li][0], LVL[ld.li][1], LVL[ld.li][2],
        LVL[ld.lo][0], LVL[ld.lo][1], LVL[ld.lo][2], ld.sd, ld.sh, ld.sw,
        ld.pd, ld.ph, ld.pw);
  }

  float* outf = (float*)d_out;
  const float headW = 1.f / 3.f;

  for (int hd = 0; hd < 3; ++hd) {
    const __hip_bfloat16* cur = xbf;
    for (int l = 0; l < 11; ++l) {
      const LD& ld = L[l];
      const float* w = (const float*)d_in[2 + (hd * 11 + l) * 3 + 0];
      const float* sc = (const float*)d_in[2 + (hd * 11 + l) * 3 + 1];
      const float* bi = (const float*)d_in[2 + (hd * 11 + l) * 3 + 2];
      const int nCh = (27 * ld.cin + 31) / 32;
      {
        int n = (ld.cout >> 4) * nCh * 512;
        k_pack_w<<<(n + 255) / 256, 256, 0, stream>>>(w, wfrag, ld.cin, ld.cout, nCh);
      }
      const int Din = LVL[ld.li][0], Hin = LVL[ld.li][1], Win = LVL[ld.li][2];
      const int Dout = LVL[ld.lo][0], Hout = LVL[ld.lo][1], Wout = LVL[ld.lo][2];
      const int nTW = (Wout + 15) / 16;
      const int tiles = Dout * Hout * nTW;
      const int blocks = (tiles + WAVES_PER_BLOCK - 1) / WAVES_PER_BLOCK;
      __hip_bfloat16* outp = (l & 1) ? actB : actA;
      const unsigned char* mp = mk[ld.lo];
      const int accumAdd = (hd > 0) ? 1 : 0;
#define CONV_ARGS                                                        \
  cur, wfrag, sc, bi, mp, outp, outf, headW, accumAdd, Din, Hin, Win,    \
      Dout, Hout, Wout, ld.sd, ld.sh, ld.sw, ld.pd, ld.ph, ld.pw
      switch (l) {
        case 0: k_conv_wmma<4, 16, false><<<blocks, 256, 0, stream>>>(CONV_ARGS); break;
        case 1: k_conv_wmma<16, 16, false><<<blocks, 256, 0, stream>>>(CONV_ARGS); break;
        case 2: k_conv_wmma<16, 32, false><<<blocks, 256, 0, stream>>>(CONV_ARGS); break;
        case 3:
        case 4: k_conv_wmma<32, 32, false><<<blocks, 256, 0, stream>>>(CONV_ARGS); break;
        case 5: k_conv_wmma<32, 64, false><<<blocks, 256, 0, stream>>>(CONV_ARGS); break;
        case 6:
        case 7:
        case 8:
        case 9: k_conv_wmma<64, 64, false><<<blocks, 256, 0, stream>>>(CONV_ARGS); break;
        case 10: k_conv_wmma<64, 64, true><<<blocks, 256, 0, stream>>>(CONV_ARGS); break;
      }
#undef CONV_ARGS
      cur = outp;
    }
  }
  (void)in_sizes; (void)n_in; (void)out_size; (void)ws_size;
}
